// LabelDictionaryAndCrossEntropy_8924942041499
// MI455X (gfx1250) — matrix operand order repaired and checked by reading, not compiled, not run on hardware
//
#include <hip/hip_runtime.h>
#include <hip/hip_bf16.h>
#include <math.h>

#define BDIM 8192
#define CDIM 1024
#define DDIM 512

typedef __attribute__((ext_vector_type(8)))  __bf16       v8bf;
typedef __attribute__((ext_vector_type(16))) __bf16       v16bf;
typedef __attribute__((ext_vector_type(8)))  float        v8f;
typedef __attribute__((ext_vector_type(4)))  unsigned int v4u;
typedef __attribute__((ext_vector_type(8)))  int          v8i;
typedef __attribute__((ext_vector_type(4)))  int          v4i;

#if defined(__has_builtin)
#if __has_builtin(__builtin_amdgcn_tensor_load_to_lds) && \
    __has_builtin(__builtin_amdgcn_s_wait_tensorcnt)
#define HAVE_TDM 1
#endif
#endif
#ifndef HAVE_TDM
#define HAVE_TDM 0
#endif

__device__ __forceinline__ unsigned short f32_to_bf16_rne(float f) {
    unsigned int u = __float_as_uint(f);
    unsigned int r = u + 0x7FFFu + ((u >> 16) & 1u);   // round-to-nearest-even
    return (unsigned short)(r >> 16);
}

// Low 32 bits of a generic pointer into LDS == the LDS byte offset
// (AMDGPU local->flat addrspacecast is {aperture_hi32, lds_offset}).
__device__ __forceinline__ unsigned lds_offset_of(const void* p) {
    return (unsigned)(unsigned long long)p;
}

#if HAVE_TDM
// ---------------------------------------------------------------------------
// Issue a TDM 2D tile load: tile_d1 rows x tile_d0 elements (2-byte elements)
// from a row-major tensor with row stride `stride0` (elements) into LDS at
// `lds_addr`, packed row-major tile_d0 elements per row.
// Descriptor bitfields per CDNA5 ISA 8.3/8.4 (D# group0/group1).
// ---------------------------------------------------------------------------
__device__ __forceinline__ void tdm_load_2d_bf16(unsigned lds_addr, const void* gptr,
                                                 unsigned tensor_d0, unsigned tensor_d1,
                                                 unsigned tile_d0, unsigned tile_d1,
                                                 unsigned stride0)
{
    unsigned long long ga = (unsigned long long)gptr;
    v4u g0;
    g0[0] = 1u;                                               // count=1, user descriptor
    g0[1] = lds_addr;                                         // lds_addr [63:32]
    g0[2] = (unsigned)(ga & 0xffffffffu);                     // global_addr lo
    g0[3] = (unsigned)((ga >> 32) & 0x01ffffffu) | (2u << 30); // addr hi, type=2
    v8i g1;
    g1[0] = 0x00010000;                                       // data_size=1 (2B), mask=0
    g1[1] = (int)((tensor_d0 & 0xffffu) << 16);               // tensor_dim0[15:0] @ [63:48]
    g1[2] = (int)(((tensor_d0 >> 16) & 0xffffu) |             // tensor_dim0[31:16]
                  ((tensor_d1 & 0xffffu) << 16));             // tensor_dim1[15:0]
    g1[3] = (int)(((tensor_d1 >> 16) & 0xffffu) |             // tensor_dim1[31:16]
                  ((tile_d0 & 0xffffu) << 16));               // tile_dim0
    g1[4] = (int)(tile_d1 & 0xffffu);                         // tile_dim1 (tile_dim2=0)
    g1[5] = (int)stride0;                                     // tensor_dim0_stride lo32
    g1[6] = 0;                                                // stride0 hi16, stride1 lo16
    g1[7] = 0;                                                // stride1 hi32
    v4i gz = {0, 0, 0, 0};
#if __clang_major__ >= 23
    v8i gz8 = {0, 0, 0, 0, 0, 0, 0, 0};
    __builtin_amdgcn_tensor_load_to_lds(g0, g1, gz, gz, gz8, 0);
#else
    __builtin_amdgcn_tensor_load_to_lds(g0, g1, gz, gz, 0);
#endif
}
#endif  // HAVE_TDM

// ---------------------------------------------------------------------------
// Kernel 0: zero the scalar output (harness poisons d_out, we accumulate into it)
// ---------------------------------------------------------------------------
__global__ void zero_out_kernel(float* out) {
    if (threadIdx.x == 0) out[0] = 0.0f;
}

// ---------------------------------------------------------------------------
// Kernel 1: per-class deterministic scatter + EMA update + L2-normalize -> bf16
// ---------------------------------------------------------------------------
__global__ __launch_bounds__(256)
void build_dict_kernel(const float* __restrict__ feat,
                       const float* __restrict__ dict_w,
                       const int*   __restrict__ targets,
                       unsigned short* __restrict__ dict_bf)
{
    __shared__ int   sT[BDIM];
    __shared__ float red[256];
    const int c   = blockIdx.x;
    const int tid = threadIdx.x;

    for (int i = tid; i < BDIM; i += 256) sT[i] = targets[i];
    __syncthreads();

    const int d0 = tid, d1 = tid + 256;
    float s0 = 0.0f, s1 = 0.0f, cnt = 0.0f;
    for (int b = 0; b < BDIM; ++b) {
        if (sT[b] == c) {
            const float* fr = feat + (size_t)b * DDIM;
            s0 += fr[d0];
            s1 += fr[d1];
            cnt += 1.0f;
        }
    }

    const float inv_c = 1.0f / (cnt + 1e-6f);
    const float valid = (cnt > 0.0f) ? 1.0f : 0.0f;
    const float* dw = dict_w + (size_t)c * DDIM;
    const float w0 = dw[d0], w1 = dw[d1];
    const float nd0 = w0 + 0.1f * ((s0 * inv_c) - w0) * valid;
    const float nd1 = w1 + 0.1f * ((s1 * inv_c) - w1) * valid;

    red[tid] = nd0 * nd0 + nd1 * nd1;
    __syncthreads();
    for (int off = 128; off > 0; off >>= 1) {
        if (tid < off) red[tid] += red[tid + off];
        __syncthreads();
    }
    const float inv_n = 1.0f / fmaxf(sqrtf(red[0]), 1e-8f);

    unsigned short* outr = dict_bf + (size_t)c * DDIM;
    outr[d0] = f32_to_bf16_rne(nd0 * inv_n);
    outr[d1] = f32_to_bf16_rne(nd1 * inv_n);
}

// ---------------------------------------------------------------------------
// Kernel 2: row-normalize feat -> bf16
// ---------------------------------------------------------------------------
__global__ __launch_bounds__(256)
void feat_norm_kernel(const float* __restrict__ feat,
                      unsigned short* __restrict__ feat_bf)
{
    __shared__ float red[256];
    const int b   = blockIdx.x;
    const int tid = threadIdx.x;
    const float* fr = feat + (size_t)b * DDIM;
    const float f0 = fr[tid], f1 = fr[tid + 256];

    red[tid] = f0 * f0 + f1 * f1;
    __syncthreads();
    for (int off = 128; off > 0; off >>= 1) {
        if (tid < off) red[tid] += red[tid + off];
        __syncthreads();
    }
    const float inv_n = 1.0f / fmaxf(sqrtf(red[0]), 1e-8f);

    unsigned short* outr = feat_bf + (size_t)b * DDIM;
    outr[tid]       = f32_to_bf16_rne(f0 * inv_n);
    outr[tid + 256] = f32_to_bf16_rne(f1 * inv_n);
}

// ---------------------------------------------------------------------------
// Per-wave compute for one K-step: preload the A fragment (16x32) and ALL 8
// B fragments (32x16 each) from LDS first, then issue the 8 WMMAs
// back-to-back, so the LDS loads overlap each other and the matrix pipe is
// not serialized on s_wait_dscnt per WMMA.
// Fragment layouts per CDNA5 ISA 7.12.2 (16-bit A 16x32, B 32x16):
//   A lane L: row (L%16); K runs [8h, +8) and [16+8h, +8), h=L/16
//   B lane L: dict row (L%16); K runs [8h, +8) and [16+8h, +8)
// ---------------------------------------------------------------------------
__device__ __forceinline__ void wave_kstep(const unsigned short* __restrict__ tA,
                                           const unsigned short* __restrict__ tB,
                                           v8f* __restrict__ acc,
                                           int wave, int half, int l16)
{
    const unsigned short* ar = tA + (wave * 16 + l16) * 32;
    const v8bf a_lo = *reinterpret_cast<const v8bf*>(ar + half * 8);
    const v8bf a_hi = *reinterpret_cast<const v8bf*>(ar + 16 + half * 8);
    v16bf a;
#pragma unroll
    for (int i = 0; i < 8; ++i) { a[i] = a_lo[i]; a[8 + i] = a_hi[i]; }

    v16bf b[8];
#pragma unroll
    for (int nt = 0; nt < 8; ++nt) {
        const unsigned short* br = tB + (nt * 16 + l16) * 32;
        const v8bf b_lo = *reinterpret_cast<const v8bf*>(br + half * 8);
        const v8bf b_hi = *reinterpret_cast<const v8bf*>(br + 16 + half * 8);
#pragma unroll
        for (int i = 0; i < 8; ++i) { b[nt][i] = b_lo[i]; b[nt][8 + i] = b_hi[i]; }
    }

#pragma unroll
    for (int nt = 0; nt < 8; ++nt) {
        acc[nt] = __builtin_amdgcn_wmma_f32_16x16x32_bf16(
            false, a, false, b[nt], (short)0, acc[nt], false, false);
    }
}

// ---------------------------------------------------------------------------
// Kernel 3: simi = feat_n @ dict_n^T.
// Block = 8 waves computes a 128x128 tile; wave w owns rows [w*16, w*16+16).
// K-steps of 32 staged through LDS. TDM (tensor_load_to_lds) double-buffered:
// wave 0 issues the next K-step's two tile DMAs before computing the current
// one, then s_wait_tensorcnt 0 + barrier publishes the buffer.
// ---------------------------------------------------------------------------
__global__ __launch_bounds__(256)
void simi_gemm_kernel(const unsigned short* __restrict__ feat_bf,
                      const unsigned short* __restrict__ dict_bf,
                      float* __restrict__ simi)
{
    __shared__ unsigned short ldsA[2][128 * 32];   // 2 x 8 KB
    __shared__ unsigned short ldsB[2][128 * 32];   // 2 x 8 KB

    const int lane   = threadIdx.x & 31;
    const int wave   = threadIdx.x >> 5;
    const int half   = lane >> 4;
    const int l16    = lane & 15;
    const int n_base = blockIdx.x * 128;
    const int m_base = blockIdx.y * 128;

    v8f acc[8] = {{}, {}, {}, {}, {}, {}, {}, {}};

#if HAVE_TDM
    const unsigned offA0 = lds_offset_of(&ldsA[0][0]);
    const unsigned offA1 = lds_offset_of(&ldsA[1][0]);
    const unsigned offB0 = lds_offset_of(&ldsB[0][0]);
    const unsigned offB1 = lds_offset_of(&ldsB[1][0]);

    if (wave == 0) {
        tdm_load_2d_bf16(offA0, feat_bf + (size_t)m_base * DDIM, DDIM, BDIM, 32, 128, DDIM);
        tdm_load_2d_bf16(offB0, dict_bf + (size_t)n_base * DDIM, DDIM, CDIM, 32, 128, DDIM);
        __builtin_amdgcn_s_wait_tensorcnt(0);
    }
    __syncthreads();

    int cur = 0;
    for (int k0 = 0; k0 < DDIM; k0 += 32) {
        const int nxt = cur ^ 1;
        if ((k0 + 32) < DDIM && wave == 0) {
            tdm_load_2d_bf16(nxt ? offA1 : offA0,
                             feat_bf + (size_t)m_base * DDIM + (k0 + 32),
                             DDIM, BDIM, 32, 128, DDIM);
            tdm_load_2d_bf16(nxt ? offB1 : offB0,
                             dict_bf + (size_t)n_base * DDIM + (k0 + 32),
                             DDIM, CDIM, 32, 128, DDIM);
        }

        wave_kstep(ldsA[cur], ldsB[cur], acc, wave, half, l16);

        if (wave == 0) __builtin_amdgcn_s_wait_tensorcnt(0);
        __syncthreads();
        cur = nxt;
    }
#else
    // Fallback: cooperative synchronous staging (256 threads, 128x32 tile,
    // thread t copies 16 elements of row t/2).
    const int tid = threadIdx.x;
    const int cr = tid >> 1, ch = tid & 1;
    for (int k0 = 0; k0 < DDIM; k0 += 32) {
        __syncthreads();
        {
            const unsigned short* sa = feat_bf + (size_t)(m_base + cr) * DDIM + k0 + ch * 16;
            const unsigned short* sb = dict_bf + (size_t)(n_base + cr) * DDIM + k0 + ch * 16;
            *reinterpret_cast<v8bf*>(&ldsA[0][cr * 32 + ch * 16])     = *reinterpret_cast<const v8bf*>(sa);
            *reinterpret_cast<v8bf*>(&ldsA[0][cr * 32 + ch * 16 + 8]) = *reinterpret_cast<const v8bf*>(sa + 8);
            *reinterpret_cast<v8bf*>(&ldsB[0][cr * 32 + ch * 16])     = *reinterpret_cast<const v8bf*>(sb);
            *reinterpret_cast<v8bf*>(&ldsB[0][cr * 32 + ch * 16 + 8]) = *reinterpret_cast<const v8bf*>(sb + 8);
        }
        __syncthreads();
        wave_kstep(ldsA[0], ldsB[0], acc, wave, half, l16);
    }
#endif

    // C/D layout: VGPR r -> row m + r + 8*half, col n + l16
#pragma unroll
    for (int nt = 0; nt < 8; ++nt) {
        float* srow = simi + (size_t)(m_base + wave * 16 + half * 8) * CDIM
                           + n_base + nt * 16 + l16;
#pragma unroll
        for (int r = 0; r < 8; ++r) srow[(size_t)r * CDIM] = acc[nt][r];
    }
}

// ---------------------------------------------------------------------------
// Kernel 4: per-row loss. Using sum(softmax)=1:
//   loss_b = 0.5*lse(pred_b) - 0.25*(sum_c e_c*pred_c)/(sum_c e_c) - 0.25*pred[b,t_b]
// ---------------------------------------------------------------------------
__global__ __launch_bounds__(256)
void loss_kernel(const float* __restrict__ pred,
                 const float* __restrict__ simi,
                 const int*   __restrict__ targets,
                 float* __restrict__ out)
{
    __shared__ float red[256];
    const int b   = blockIdx.x;
    const int tid = threadIdx.x;
    const float* pr = pred + (size_t)b * CDIM;
    const float* sr = simi + (size_t)b * CDIM;

    float pv[4], sv[4];
#pragma unroll
    for (int j = 0; j < 4; ++j) {
        pv[j] = pr[tid + 256 * j];
        sv[j] = sr[tid + 256 * j];
    }

    red[tid] = fmaxf(fmaxf(pv[0], pv[1]), fmaxf(pv[2], pv[3]));
    __syncthreads();
    for (int off = 128; off > 0; off >>= 1) {
        if (tid < off) red[tid] = fmaxf(red[tid], red[tid + off]);
        __syncthreads();
    }
    const float pmax = red[0];
    __syncthreads();

    float ps = 0.0f;
#pragma unroll
    for (int j = 0; j < 4; ++j) ps += expf(pv[j] - pmax);
    red[tid] = ps;
    __syncthreads();
    for (int off = 128; off > 0; off >>= 1) {
        if (tid < off) red[tid] += red[tid + off];
        __syncthreads();
    }
    const float psum = red[0];
    __syncthreads();

    red[tid] = fmaxf(fmaxf(sv[0], sv[1]), fmaxf(sv[2], sv[3]));
    __syncthreads();
    for (int off = 128; off > 0; off >>= 1) {
        if (tid < off) red[tid] = fmaxf(red[tid], red[tid + off]);
        __syncthreads();
    }
    const float smax = red[0];
    __syncthreads();

    float es = 0.0f, ws = 0.0f;
#pragma unroll
    for (int j = 0; j < 4; ++j) {
        const float e = expf(sv[j] - smax);
        es += e;
        ws += e * pv[j];
    }
    red[tid] = es;
    __syncthreads();
    for (int off = 128; off > 0; off >>= 1) {
        if (tid < off) red[tid] += red[tid + off];
        __syncthreads();
    }
    const float S = red[0];
    __syncthreads();

    red[tid] = ws;
    __syncthreads();
    for (int off = 128; off > 0; off >>= 1) {
        if (tid < off) red[tid] += red[tid + off];
        __syncthreads();
    }
    const float W = red[0];

    if (tid == 0) {
        const float lse    = pmax + logf(psum);
        const int   t      = targets[b];
        const float loss_b = 0.5f * lse - 0.25f * (W / S) - 0.25f * pr[t];
        atomicAdd(out, loss_b * (1.0f / (float)BDIM));
    }
}

// ---------------------------------------------------------------------------
extern "C" void kernel_launch(void* const* d_in, const int* in_sizes, int n_in,
                              void* d_out, int out_size, void* d_ws, size_t ws_size,
                              hipStream_t stream)
{
    (void)in_sizes; (void)n_in; (void)out_size; (void)ws_size;
    const float* pred    = (const float*)d_in[0];
    const float* feat    = (const float*)d_in[1];
    const float* dict_w  = (const float*)d_in[2];
    const int*   targets = (const int*)d_in[3];
    float* out = (float*)d_out;

    char* ws = (char*)d_ws;
    unsigned short* dict_bf = (unsigned short*)(ws);                                   // 1 MB
    unsigned short* feat_bf = (unsigned short*)(ws + (size_t)CDIM * DDIM * 2);         // 8 MB
    float*          simi    = (float*)(ws + (size_t)CDIM * DDIM * 2
                                          + (size_t)BDIM * DDIM * 2);                  // 32 MB

    zero_out_kernel<<<1, 64, 0, stream>>>(out);
    build_dict_kernel<<<CDIM, 256, 0, stream>>>(feat, dict_w, targets, dict_bf);
    feat_norm_kernel<<<BDIM, 256, 0, stream>>>(feat, feat_bf);
    simi_gemm_kernel<<<dim3(CDIM / 128, BDIM / 128), 256, 0, stream>>>(feat_bf, dict_bf, simi);
    loss_kernel<<<BDIM, 256, 0, stream>>>(pred, simi, targets, out);
}
